// AttnHead_3315714752638
// MI455X (gfx1250) — compile-verified, matrix operand order checked by reading, not run on hardware
//
#include <hip/hip_runtime.h>

#define B_ 2
#define N_ 6144
#define H_ 256
#define O_ 128

typedef __attribute__((ext_vector_type(2)))  float   v2f;
typedef __attribute__((ext_vector_type(8)))  float   v8f;
typedef __attribute__((ext_vector_type(16))) __bf16  v16bf;
typedef __attribute__((ext_vector_type(4)))  unsigned int v4u;
typedef __attribute__((ext_vector_type(8)))  int     v8i_;
typedef __attribute__((ext_vector_type(4)))  int     v4i_;

static __device__ __forceinline__ unsigned short f2bf(float x) {
    unsigned u = __float_as_uint(x);
    u += 0x7fffu + ((u >> 16) & 1u);          // round-to-nearest-even
    return (unsigned short)(u >> 16);
}

// ---------------------------------------------------------------------------
// TDM: DMA a 128(o) x 32(j) bf16 tile (rows stride N_ elements in global) into
// LDS at lds_off, with hardware padding of +4 DWORDs every 16 DWORDs
// => LDS row stride = 32 bf16 data + 8 bf16 pad = 40 (bank-conflict free).
// D# per CDNA5 ISA 8.3/8.4 (2D tensor; groups 2/3 zero, tile_dim2=0).
// Toolchain here is amdgpu-toolchain (clang-23): 6-arg builtin form.
// ---------------------------------------------------------------------------
static __device__ __forceinline__ void tdm_load_tile(const unsigned short* gptr,
                                                     unsigned lds_off) {
    const unsigned long long ga = (unsigned long long)(size_t)gptr;
    v4u g0;
    g0.x = 1u;                                           // count=1, no gather
    g0.y = lds_off;                                      // lds_addr (bytes)
    g0.z = (unsigned)(ga & 0xffffffffu);                 // global_addr[31:0]
    g0.w = (unsigned)((ga >> 32) & 0x01ffffffu)          // global_addr[56:32]
         | (2u << 30);                                   // type=2 ("image")
    v8i_ g1;
    g1[0] = (int)((1u << 16)     // data_size = 1 -> 2 bytes
                | (1u << 20)     // pad_enable
                | (3u << 22)     // pad_interval = 3 -> every 16 DWORDs (64B row)
                | (3u << 25));   // pad_amount   = 3 -> 4 DWORDs (8 bf16)
    g1[1] = (int)((unsigned)N_ << 16);   // tensor_dim0[15:0] = 6144
    g1[2] = (int)(128u << 16);           // tensor_dim0 hi=0 | tensor_dim1 = 128
    g1[3] = (int)(32u  << 16);           // tensor_dim1 hi=0 | tile_dim0   = 32
    g1[4] = 128;                         // tile_dim1 = 128 | tile_dim2 = 0
    g1[5] = N_;                          // tensor_dim0_stride lo = 6144
    g1[6] = 0;                           // stride hi | tensor_dim1_stride (2D: unused)
    g1[7] = 0;
    const v4i_ z4 = {0, 0, 0, 0};
    const v8i_ z8 = {0, 0, 0, 0, 0, 0, 0, 0};
    __builtin_amdgcn_tensor_load_to_lds(g0, g1, z4, z4, z8, 0);
}

// ---------------------------------------------------------------------------
// Kernel 1: seq_fts[b,n,o] = sum_h seq[b,n,h] * W_fts[o,h]   (fp32 WMMA)
// Writes fp32 [b][n][o] and transposed bf16 [b][o][n].
// ---------------------------------------------------------------------------
__global__ __launch_bounds__(128) void proj_kernel(
    const float* __restrict__ seq, const float* __restrict__ Wf,
    float* __restrict__ sf, unsigned short* __restrict__ sfT)
{
    const int b  = blockIdx.y;
    const int n0 = blockIdx.x * 16;
    const int w  = threadIdx.x >> 5;
    const int l  = threadIdx.x & 31;
    const int lr = l & 15, lh = l >> 4;

    const float* arow  = seq + ((size_t)(b * N_ + n0 + lr)) * H_ + 2 * lh;
    const float* brow0 = Wf + (size_t)(w * 32 + lr) * H_ + 2 * lh;
    const float* brow1 = brow0 + 16 * H_;

    v8f c0 = (v8f)(0.0f);
    v8f c1 = (v8f)(0.0f);

    #pragma unroll 4
    for (int k0 = 0; k0 < H_; k0 += 4) {
        v2f a  = *(const v2f*)(arow  + k0);
        v2f bp = *(const v2f*)(brow0 + k0);
        v2f bq = *(const v2f*)(brow1 + k0);
        c0 = __builtin_amdgcn_wmma_f32_16x16x4_f32(false, a, false, bp, (short)0, c0, false, false);
        c1 = __builtin_amdgcn_wmma_f32_16x16x4_f32(false, a, false, bq, (short)0, c1, false, false);
    }

    const int ob = w * 32;
    #pragma unroll
    for (int r = 0; r < 8; ++r) {
        const int n = n0 + r + 8 * lh;
        const size_t base = ((size_t)(b * N_) + n) * O_;
        const float x0 = c0[r], x1 = c1[r];
        sf[base + ob + lr]      = x0;
        sf[base + ob + 16 + lr] = x1;
        sfT[((size_t)b * O_ + ob + lr)      * N_ + n] = f2bf(x0);
        sfT[((size_t)b * O_ + ob + 16 + lr) * N_ + n] = f2bf(x1);
    }
}

// ---------------------------------------------------------------------------
// Kernel 2: f1/f2 row dots, one wave32 per row, shuffle reduction.
// ---------------------------------------------------------------------------
__global__ __launch_bounds__(256) void fvec_kernel(
    const float* __restrict__ sf,
    const float* __restrict__ w1, const float* __restrict__ b1,
    const float* __restrict__ w2, const float* __restrict__ b2,
    float* __restrict__ f1, float* __restrict__ f2)
{
    const int w = threadIdx.x >> 5, l = threadIdx.x & 31;
    const int row = blockIdx.x * 8 + w;
    const float4 v  = *(const float4*)(sf + (size_t)row * O_ + l * 4);
    const float4 a1 = *(const float4*)(w1 + l * 4);
    const float4 a2 = *(const float4*)(w2 + l * 4);
    float s1 = v.x * a1.x + v.y * a1.y + v.z * a1.z + v.w * a1.w;
    float s2 = v.x * a2.x + v.y * a2.y + v.z * a2.z + v.w * a2.w;
    #pragma unroll
    for (int off = 16; off > 0; off >>= 1) {
        s1 += __shfl_xor(s1, off, 32);
        s2 += __shfl_xor(s2, off, 32);
    }
    if (l == 0) { f1[row] = s1 + b1[0]; f2[row] = s2 + b2[0]; }
}

// ---------------------------------------------------------------------------
// Kernel 3: fused sigmoid-softmax + GEMM2 + ELU, TDM double-buffered.
// ---------------------------------------------------------------------------
__global__ __launch_bounds__(128) void attn_kernel(
    const unsigned short* __restrict__ sfT,
    const float* __restrict__ f1v, const float* __restrict__ f2v,
    const float* __restrict__ bias, float* __restrict__ out)
{
    __shared__ unsigned short lds[2][O_ * 40];   // double-buffered [o][j], stride 40

    const int b  = blockIdx.y;
    const int i0 = blockIdx.x * 64;
    const int w  = threadIdx.x >> 5;
    const int l  = threadIdx.x & 31;
    const int lr = l & 15, lh = l >> 4;
    const bool issuer = (w == 0);

    const int i = i0 + 16 * w + lr;
    const float F1b = f1v[b * N_ + i];
    const float F1o = f1v[(1 - b) * N_ + i];
    const float* p2b = f2v + b * N_;
    const float* p2o = f2v + (1 - b) * N_;
    const unsigned short* gsrc = sfT + (size_t)b * O_ * N_;

    v8f acc[8];
    #pragma unroll
    for (int t = 0; t < 8; ++t) acc[t] = (v8f)(0.0f);

    // prologue: DMA first tile into buffer 0
    if (issuer) tdm_load_tile(gsrc, (unsigned)(size_t)&lds[0][0]);

    for (int k = 0; k < N_ / 32; ++k) {
        const int j0 = k * 32;
        if (issuer) __builtin_amdgcn_s_wait_tensorcnt(0);  // tile k landed in LDS
        __syncthreads();                                    // visible to all waves
        if (issuer && (k + 1) < (N_ / 32))                  // overlap: DMA tile k+1
            tdm_load_tile(gsrc + j0 + 32, (unsigned)(size_t)&lds[(k + 1) & 1][0]);

        const unsigned short* buf = &lds[k & 1][0];

        // ---- A fragment: 16 sigmoid attention coefficients per lane ----
        const int jA = j0 + 8 * lh;          // halves 0..7  -> j = jA+0..7
        float fb[16], fo[16];                // halves 8..15 -> j = jA+16..23
        *(float4*)&fb[0]  = *(const float4*)(p2b + jA);
        *(float4*)&fb[4]  = *(const float4*)(p2b + jA + 4);
        *(float4*)&fb[8]  = *(const float4*)(p2b + jA + 16);
        *(float4*)&fb[12] = *(const float4*)(p2b + jA + 20);
        *(float4*)&fo[0]  = *(const float4*)(p2o + jA);
        *(float4*)&fo[4]  = *(const float4*)(p2o + jA + 4);
        *(float4*)&fo[8]  = *(const float4*)(p2o + jA + 16);
        *(float4*)&fo[12] = *(const float4*)(p2o + jA + 20);

        union { v16bf v; unsigned short u[16]; } A;
        #pragma unroll
        for (int h = 0; h < 16; ++h) {
            const float xb = F1b + fb[h];
            const float xo = F1o + fo[h];
            const float gb = fmaxf(xb, 0.01f * xb);   // leaky_relu (slope < 1)
            const float go = fmaxf(xo, 0.01f * xo);
            const float p  = __builtin_amdgcn_rcpf(1.0f + __expf(go - gb));
            A.u[h] = f2bf(p);
        }

        // ---- 8 WMMA tiles across O ----
        #pragma unroll
        for (int ot = 0; ot < 8; ++ot) {
            union { v16bf v; uint4 q[2]; } Bf;
            const int o = ot * 16 + lr;
            Bf.q[0] = *(const uint4*)&buf[o * 40 + 8 * lh];        // K = 8*lh+0..7
            Bf.q[1] = *(const uint4*)&buf[o * 40 + 8 * lh + 16];   // K = 16+8*lh+0..7
            acc[ot] = __builtin_amdgcn_wmma_f32_16x16x32_bf16(
                false, A.v, false, Bf.v, (short)0, acc[ot], false, false);
        }
    }

    // ---- epilogue: bias + ELU, D layout: D[r + 8*lh][lr] ----
    const float bs = bias[0];
    #pragma unroll
    for (int ot = 0; ot < 8; ++ot) {
        #pragma unroll
        for (int r = 0; r < 8; ++r) {
            const float x = acc[ot][r] + bs;
            const float y = x > 0.0f ? x : (__expf(x) - 1.0f);
            out[((size_t)b * N_ + (i0 + 16 * w + r + 8 * lh)) * O_ + ot * 16 + lr] = y;
        }
    }
}

// ---------------------------------------------------------------------------
extern "C" void kernel_launch(void* const* d_in, const int* in_sizes, int n_in,
                              void* d_out, int out_size, void* d_ws, size_t ws_size,
                              hipStream_t stream)
{
    const float* seq  = (const float*)d_in[0];
    const float* Wf   = (const float*)d_in[1];
    const float* f1w  = (const float*)d_in[2];
    const float* f1b  = (const float*)d_in[3];
    const float* f2w  = (const float*)d_in[4];
    const float* f2b  = (const float*)d_in[5];
    const float* bias = (const float*)d_in[6];

    char* ws = (char*)d_ws;
    float* sf           = (float*)ws;                              // B*N*O f32   (6291456 B)
    float* f1           = (float*)(ws + 6291456);                  // B*N   f32   (49152 B)
    float* f2           = (float*)(ws + 6291456 + 49152);          // B*N   f32   (49152 B)
    unsigned short* sfT = (unsigned short*)(ws + 6291456 + 98304); // B*O*N bf16  (3145728 B)

    proj_kernel<<<dim3(N_ / 16, B_), 128, 0, stream>>>(seq, Wf, sf, sfT);
    fvec_kernel<<<dim3(B_ * N_ / 8), 256, 0, stream>>>(sf, f1w, f1b, f2w, f2b, f1, f2);
    attn_kernel<<<dim3(N_ / 64, B_), 128, 0, stream>>>(sfT, f1, f2, bias, (float*)d_out);
}
